// DanQCat_attn_61057255080058
// MI455X (gfx1250) — matrix operand order repaired and checked by reading, not compiled, not run on hardware
//
#include <hip/hip_runtime.h>

#define LDF 40  // LDS row stride in halfwords: keeps 16B alignment, staggers banks

typedef __attribute__((ext_vector_type(16))) __bf16 v16bf;
typedef __attribute__((ext_vector_type(8)))  float  v8f;

union FragU {
  v16bf v;
  unsigned short u[16];
  uint4 q[2];
};

__device__ __forceinline__ unsigned short f32_to_bf16(float f) {
  unsigned int x = __float_as_uint(f);
  unsigned int r = x + 0x7FFFu + ((x >> 16) & 1u);  // round-to-nearest-even
  return (unsigned short)(r >> 16);
}

__device__ __forceinline__ unsigned int pack_bf16x2(float a, float b) {
  return (unsigned int)f32_to_bf16(a) | ((unsigned int)f32_to_bf16(b) << 16);
}

// A fragment: 16x32 bf16 (M x K), LDS tile is [16][LDF] m-major.
__device__ __forceinline__ v16bf load_A_frag(const unsigned short* A, int lane) {
  const int m  = lane & 15;
  const int kb = ((lane >> 4) & 1) << 3;
  FragU f;
  f.q[0] = *(const uint4*)(A + m * LDF + kb);
  f.q[1] = *(const uint4*)(A + m * LDF + kb + 16);
  return f.v;
}

// B fragment: 32x16 bf16 (K x N), LDS tile is [16][LDF] n-major (B^T rows).
__device__ __forceinline__ v16bf load_B_frag(const unsigned short* Bt, int lane) {
  const int n  = lane & 15;
  const int kb = ((lane >> 4) & 1) << 3;
  FragU f;
  f.q[0] = *(const uint4*)(Bt + n * LDF + kb);
  f.q[1] = *(const uint4*)(Bt + n * LDF + kb + 16);
  return f.v;
}

__device__ __forceinline__ v8f wmma_bf16(v16bf a, v16bf b, v8f c) {
  return __builtin_amdgcn_wmma_f32_16x16x32_bf16(false, a, false, b, (short)0, c,
                                                 false, false);
}

__device__ __forceinline__ v8f zero8() {
  v8f z;
#pragma unroll
  for (int i = 0; i < 8; ++i) z[i] = 0.f;
  return z;
}

__device__ __forceinline__ float fast_sigmoid(float x) {
  float cx = fminf(fmaxf(x, -30.f), 30.f);
  return 1.f / (1.f + __expf(-cx));
}
__device__ __forceinline__ float fast_tanh(float x) {
  float cx = fminf(fmaxf(x, -15.f), 15.f);
  float e  = __expf(2.f * cx);
  return (e - 1.f) / (e + 1.f);
}

// ---------------------------------------------------------------------------
// Kernel 1: Conv1d(4->320,k=26) + bias + ReLU + left-pad-14 + MaxPool(13)
// Grid (5 ch-groups of 64, 3 t-blocks, 800 batch), 128 threads (4 waves).
// GEMM: 64ch x 256pos (16 N-subtiles; positions >=195 are unused padding so every
// wave owns exactly 4 subtiles) x K=104->128. B fragments are loaded once per
// k-step and reused across the 4 channel tiles, amortizing the im2col gather.
// Pools 15 windows/block and writes seq bf16 [45][800][320].
// ---------------------------------------------------------------------------
__global__ __launch_bounds__(128) void conv_pool_kernel(
    const float* __restrict__ x, const float* __restrict__ cw,
    const float* __restrict__ cb, unsigned short* __restrict__ seq) {
  __shared__ __align__(16) unsigned short Asm[64 * LDF];
  __shared__ __align__(16) unsigned short Bsm[256 * LDF];
  __shared__ float convs[64 * 256];
  __shared__ float biasS[64];

  const int cg = blockIdx.x, tb = blockIdx.y, b = blockIdx.z;
  const int c0 = cg * 64;
  const int S  = 195 * tb - 14;  // conv position of local p=0 (negative: pad)
  const int tid = threadIdx.x, lane = tid & 31;
  const int wv = __builtin_amdgcn_readfirstlane(tid >> 5);
  if (tid < 64) biasS[tid] = cb[c0 + tid];

  v8f acc[4][4];
#pragma unroll
  for (int ct = 0; ct < 4; ++ct)
#pragma unroll
    for (int j = 0; j < 4; ++j) acc[ct][j] = zero8();

  const float* xb = x + b * 2400;  // [4][600]
  for (int ks = 0; ks < 4; ++ks) {
    const int k0 = ks * 32;
    // stage A: 64 rows x 8 chunks-of-4 (aligned float4; 104 % 4 == 0 so chunks
    // are never partial)
    for (int cix = tid; cix < 512; cix += 128) {
      int r = cix >> 3, kc = cix & 7;
      int kg0 = k0 + kc * 4;
      uint2 u{0, 0};
      if (kg0 < 104) {
        float4 f = *(const float4*)(cw + (c0 + r) * 104 + kg0);
        u.x = pack_bf16x2(f.x, f.y);
        u.y = pack_bf16x2(f.z, f.w);
      }
      *(uint2*)(&Asm[r * LDF + kc * 4]) = u;
    }
    // stage B (im2col gather): 256 pos x 8 chunks-of-4
    for (int cix = tid; cix < 2048; cix += 128) {
      int p = cix >> 3, kc = cix & 7;
      int kg0 = k0 + kc * 4;
      int P = S + p;
      uint2 u{0, 0};
      if (P >= 0 && P < 575 && kg0 < 104) {
        float f[4];
#pragma unroll
        for (int z = 0; z < 4; ++z) {
          int kg = kg0 + z;
          int ic = kg / 26, kk = kg - ic * 26;
          f[z] = xb[ic * 600 + P + kk];
        }
        u.x = pack_bf16x2(f[0], f[1]);
        u.y = pack_bf16x2(f[2], f[3]);
      }
      *(uint2*)(&Bsm[p * LDF + kc * 4]) = u;
    }
    __syncthreads();
    v16bf bb[4];
#pragma unroll
    for (int j = 0; j < 4; ++j)
      bb[j] = load_B_frag(Bsm + (wv * 4 + j) * 16 * LDF, lane);
#pragma unroll
    for (int ct = 0; ct < 4; ++ct) {
      v16bf a = load_A_frag(Asm + ct * 16 * LDF, lane);
#pragma unroll
      for (int j = 0; j < 4; ++j) acc[ct][j] = wmma_bf16(a, bb[j], acc[ct][j]);
    }
    __syncthreads();
  }

  // write conv tile (+bias) to LDS: wave wv owns subtiles 4wv..4wv+3
  {
    int hi = (lane >> 4) & 1, nn = lane & 15;
#pragma unroll
    for (int ct = 0; ct < 4; ++ct) {
#pragma unroll
      for (int j = 0; j < 4; ++j) {
        int st = wv * 4 + j;
#pragma unroll
        for (int r = 0; r < 8; ++r) {
          int ch = ct * 16 + r + 8 * hi;
          convs[ch * 256 + st * 16 + nn] = acc[ct][j][r] + biasS[ch];
        }
      }
    }
  }
  __syncthreads();

  // pool: 64 channels x 15 windows (only p<=194 read; padding never touched).
  // pad positions contribute 0 and ReLU is implicit since max starts at 0.
  for (int e = tid; e < 960; e += 128) {
    int ch = e / 15, ti = e - (e / 15) * 15;
    int t = tb * 15 + ti;
    int pbase = 13 * ti;
    float m = 0.f;
#pragma unroll
    for (int w = 0; w < 13; ++w) {
      int P = S + pbase + w;
      if (P >= 0 && P < 575) m = fmaxf(m, convs[ch * 256 + pbase + w]);
    }
    seq[(t * 800 + b) * 320 + c0 + ch] = f32_to_bf16(m);
  }
}

// ---------------------------------------------------------------------------
// Kernel 2: fused single-step LSTM (both directions).
// gates = seq @ w_ih^T + b_ih + b_hh; h = sig(o)*tanh(sig(i)*tanh(g))
// f-gate skipped (c0 = 0). Grid (10 u-tiles, 563 n-tiles, 2 dirs), 128 thr.
// Writes h f32 [T*B][320] (fwd units 0..159, bwd 160..319).
// ---------------------------------------------------------------------------
__global__ __launch_bounds__(128) void lstm_gates_kernel(
    const unsigned short* __restrict__ seq,
    const float* __restrict__ wihf, const float* __restrict__ bihf,
    const float* __restrict__ bhhf, const float* __restrict__ wihb,
    const float* __restrict__ bihb, const float* __restrict__ bhhb,
    float* __restrict__ h) {
  __shared__ __align__(16) unsigned short Asm[48 * LDF];
  __shared__ __align__(16) unsigned short Bsm[64 * LDF];
  __shared__ float bsum[3][16];

  const int u0  = blockIdx.x * 16;
  const int n0  = blockIdx.y * 64;
  const int dir = blockIdx.z;
  const float* wih = dir ? wihb : wihf;
  const float* bih = dir ? bihb : bihf;
  const float* bhh = dir ? bhhb : bhhf;
  const int tid = threadIdx.x, lane = tid & 31;
  const int wv = __builtin_amdgcn_readfirstlane(tid >> 5);

  if (tid < 48) {
    int g = tid >> 4, r = tid & 15;
    int gb = (g == 0) ? 0 : (g == 1 ? 320 : 480);  // i, g, o row bases
    int row = gb + u0 + r;
    bsum[g][r] = bih[row] + bhh[row];
  }

  v8f ci = zero8(), cg = zero8(), co = zero8();
  for (int ks = 0; ks < 10; ++ks) {
    const int k0 = ks * 32;
    // stage A: 48 rows x 8 chunks-of-4 (w_ih rows are 320 f32 -> 16B aligned)
    for (int cix = tid; cix < 384; cix += 128) {
      int rr = cix >> 3, kc = cix & 7;
      int g = rr >> 4, r = rr & 15;
      int gb = (g == 0) ? 0 : (g == 1 ? 320 : 480);
      float4 f = *(const float4*)(wih + (gb + u0 + r) * 320 + k0 + kc * 4);
      uint2 u;
      u.x = pack_bf16x2(f.x, f.y);
      u.y = pack_bf16x2(f.z, f.w);
      *(uint2*)(&Asm[rr * LDF + kc * 4]) = u;
    }
    // stage B: 64 rows x 4 uint4 chunks (seq rows are 640B -> 16B aligned)
    for (int cix = tid; cix < 256; cix += 128) {
      int nn = cix >> 2, q = cix & 3;
      int n = n0 + nn;
      uint4 v{0, 0, 0, 0};
      if (n < 36000) v = *(const uint4*)(seq + n * 320 + k0 + q * 8);
      *(uint4*)(&Bsm[nn * LDF + q * 8]) = v;
    }
    __syncthreads();
    v16bf bb = load_B_frag(Bsm + wv * 16 * LDF, lane);
    v16bf ai = load_A_frag(Asm, lane);
    ci = wmma_bf16(ai, bb, ci);
    v16bf ag = load_A_frag(Asm + 16 * LDF, lane);
    cg = wmma_bf16(ag, bb, cg);
    v16bf ao = load_A_frag(Asm + 32 * LDF, lane);
    co = wmma_bf16(ao, bb, co);
    __syncthreads();
  }

  const int hi = (lane >> 4) & 1;
  const int n = n0 + wv * 16 + (lane & 15);
  if (n < 36000) {
#pragma unroll
    for (int r = 0; r < 8; ++r) {
      int ur = r + 8 * hi;
      float gi = ci[r] + bsum[0][ur];
      float gg = cg[r] + bsum[1][ur];
      float go = co[r] + bsum[2][ur];
      float c  = fast_sigmoid(gi) * fast_tanh(gg);
      float hv = fast_sigmoid(go) * fast_tanh(c);
      h[n * 320 + dir * 160 + u0 + ur] = hv;
    }
  }
}

// ---------------------------------------------------------------------------
// Kernel 3: gene = geneexpr @ gene_w^T + gene_b  ->  gene_pre f32 [320][800]
// Grid (20 m-tiles, 13 n-tiles), 128 threads. K = 19795 (tail zero-padded;
// rows are odd-strided so loads stay scalar, stores are packed b64).
// ---------------------------------------------------------------------------
__global__ __launch_bounds__(128) void gene_gemm_kernel(
    const float* __restrict__ gw, const float* __restrict__ ge,
    const float* __restrict__ gb, float* __restrict__ gene_pre) {
  __shared__ __align__(16) unsigned short Asm[16 * LDF];
  __shared__ __align__(16) unsigned short Bsm[64 * LDF];
  const int m0 = blockIdx.x * 16, n0 = blockIdx.y * 64;
  const int tid = threadIdx.x, lane = tid & 31;
  const int wv = __builtin_amdgcn_readfirstlane(tid >> 5);
  v8f c = zero8();
  const int KS = (19795 + 31) / 32;  // 619
  for (int ks = 0; ks < KS; ++ks) {
    const int k0 = ks * 32;
    for (int cix = tid; cix < 128; cix += 128) {
      int r = cix >> 3, kc = cix & 7;
      int kg0 = k0 + kc * 4;
      const float* src = gw + (size_t)(m0 + r) * 19795;
      float f0 = (kg0 + 0 < 19795) ? src[kg0 + 0] : 0.f;
      float f1 = (kg0 + 1 < 19795) ? src[kg0 + 1] : 0.f;
      float f2 = (kg0 + 2 < 19795) ? src[kg0 + 2] : 0.f;
      float f3 = (kg0 + 3 < 19795) ? src[kg0 + 3] : 0.f;
      uint2 u;
      u.x = pack_bf16x2(f0, f1);
      u.y = pack_bf16x2(f2, f3);
      *(uint2*)(&Asm[r * LDF + kc * 4]) = u;
    }
    for (int cix = tid; cix < 512; cix += 128) {
      int nn = cix >> 3, kc = cix & 7;
      int kg0 = k0 + kc * 4;
      int n = n0 + nn;
      uint2 u{0, 0};
      if (n < 800) {
        const float* src = ge + (size_t)n * 19795;
        float f0 = (kg0 + 0 < 19795) ? src[kg0 + 0] : 0.f;
        float f1 = (kg0 + 1 < 19795) ? src[kg0 + 1] : 0.f;
        float f2 = (kg0 + 2 < 19795) ? src[kg0 + 2] : 0.f;
        float f3 = (kg0 + 3 < 19795) ? src[kg0 + 3] : 0.f;
        u.x = pack_bf16x2(f0, f1);
        u.y = pack_bf16x2(f2, f3);
      }
      *(uint2*)(&Bsm[nn * LDF + kc * 4]) = u;
    }
    __syncthreads();
    v16bf a = load_A_frag(Asm, lane);
    v16bf b = load_B_frag(Bsm + wv * 16 * LDF, lane);
    c = wmma_bf16(a, b, c);
    __syncthreads();
  }
  const int hi = (lane >> 4) & 1;
  const int n = n0 + wv * 16 + (lane & 15);
  if (n < 800) {
#pragma unroll
    for (int r = 0; r < 8; ++r) {
      int m = m0 + r + 8 * hi;
      gene_pre[m * 800 + n] = c[r] + gb[m];
    }
  }
}

// ---------------------------------------------------------------------------
// Kernel 4: BatchNorm1d (training forward, biased var) + ReLU, transpose to
// gene_norm f32 [800][320]. One block per channel.
// ---------------------------------------------------------------------------
__global__ __launch_bounds__(256) void bn_kernel(
    const float* __restrict__ gene_pre, const float* __restrict__ gamma,
    const float* __restrict__ beta, float* __restrict__ gene_norm) {
  __shared__ float red[256], red2[256];
  const int m = blockIdx.x, tid = threadIdx.x;
  float s = 0.f, s2 = 0.f;
  for (int n = tid; n < 800; n += 256) {
    float v = gene_pre[m * 800 + n];
    s += v;
    s2 += v * v;
  }
  red[tid] = s;
  red2[tid] = s2;
  __syncthreads();
  for (int off = 128; off > 0; off >>= 1) {
    if (tid < off) {
      red[tid] += red[tid + off];
      red2[tid] += red2[tid + off];
    }
    __syncthreads();
  }
  float mean  = red[0] * (1.f / 800.f);
  float var   = red2[0] * (1.f / 800.f) - mean * mean;
  float scale = gamma[m] * rsqrtf(var + 1e-5f);
  float shift = beta[m] - mean * scale;
  for (int n = tid; n < 800; n += 256) {
    float v = gene_pre[m * 800 + n] * scale + shift;
    gene_norm[n * 320 + m] = v > 0.f ? v : 0.f;
  }
}

// ---------------------------------------------------------------------------
// Kernel 5: attn[t,b] = <h[t,b,:], gene[b,:]>; flat[b, t*320+d] = attn*h.
// Grid (800, 45), 128 threads. flat written bf16 (lin-GEMM B operand).
// ---------------------------------------------------------------------------
__global__ __launch_bounds__(128) void attn_kernel(
    const float* __restrict__ h, const float* __restrict__ gene_norm,
    unsigned short* __restrict__ flat) {
  __shared__ float red[128];
  const int b = blockIdx.x, t = blockIdx.y, tid = threadIdx.x;
  const float* hp = h + (t * 800 + b) * 320;
  const float* gp = gene_norm + b * 320;
  float s = 0.f;
  for (int d = tid; d < 320; d += 128) s += hp[d] * gp[d];
  red[tid] = s;
  __syncthreads();
  for (int off = 64; off > 0; off >>= 1) {
    if (tid < off) red[tid] += red[tid + off];
    __syncthreads();
  }
  float attn = red[0];
  for (int d = tid; d < 320; d += 128)
    flat[b * 14400 + t * 320 + d] = f32_to_bf16(attn * hp[d]);
}

// ---------------------------------------------------------------------------
// Kernel 6: hid = ReLU(flat @ lin_w^T + lin_b) -> f32 [800][925]
// Grid (58 m-tiles (925->928), 13 n-tiles), 128 threads, K = 14400.
// ---------------------------------------------------------------------------
__global__ __launch_bounds__(128) void lin_gemm_kernel(
    const float* __restrict__ lw, const unsigned short* __restrict__ flat,
    const float* __restrict__ lb, float* __restrict__ hid) {
  __shared__ __align__(16) unsigned short Asm[16 * LDF];
  __shared__ __align__(16) unsigned short Bsm[64 * LDF];
  const int m0 = blockIdx.x * 16, n0 = blockIdx.y * 64;
  const int tid = threadIdx.x, lane = tid & 31;
  const int wv = __builtin_amdgcn_readfirstlane(tid >> 5);
  v8f c = zero8();
  for (int ks = 0; ks < 450; ++ks) {
    const int k0 = ks * 32;
    // stage A: 16 rows x 8 chunks-of-4 (lin_w rows 14400 f32 -> 16B aligned)
    for (int cix = tid; cix < 128; cix += 128) {
      int r = cix >> 3, kc = cix & 7;
      int m = m0 + r;
      uint2 u{0, 0};
      if (m < 925) {
        float4 f = *(const float4*)(lw + (size_t)m * 14400 + k0 + kc * 4);
        u.x = pack_bf16x2(f.x, f.y);
        u.y = pack_bf16x2(f.z, f.w);
      }
      *(uint2*)(&Asm[r * LDF + kc * 4]) = u;
    }
    // stage B: 64 rows x 4 uint4 chunks (flat rows 28800B -> 16B aligned)
    for (int cix = tid; cix < 256; cix += 128) {
      int nn = cix >> 2, q = cix & 3;
      int n = n0 + nn;
      uint4 v{0, 0, 0, 0};
      if (n < 800) v = *(const uint4*)(flat + (size_t)n * 14400 + k0 + q * 8);
      *(uint4*)(&Bsm[nn * LDF + q * 8]) = v;
    }
    __syncthreads();
    v16bf a = load_A_frag(Asm, lane);
    v16bf b = load_B_frag(Bsm + wv * 16 * LDF, lane);
    c = wmma_bf16(a, b, c);
    __syncthreads();
  }
  const int hi = (lane >> 4) & 1;
  const int n = n0 + wv * 16 + (lane & 15);
  if (n < 800) {
#pragma unroll
    for (int r = 0; r < 8; ++r) {
      int m = m0 + r + 8 * hi;
      if (m < 925) {
        float v = c[r] + lb[m];
        hid[n * 925 + m] = v > 0.f ? v : 0.f;
      }
    }
  }
}

// ---------------------------------------------------------------------------
// Kernel 7: out[b] = <hid[b,:], out_w> + out_b. One block per batch row.
// ---------------------------------------------------------------------------
__global__ __launch_bounds__(128) void out_kernel(
    const float* __restrict__ hid, const float* __restrict__ ow,
    const float* __restrict__ ob, float* __restrict__ out) {
  __shared__ float red[128];
  const int b = blockIdx.x, tid = threadIdx.x;
  float s = 0.f;
  for (int d = tid; d < 925; d += 128) s += hid[b * 925 + d] * ow[d];
  red[tid] = s;
  __syncthreads();
  for (int off = 64; off > 0; off >>= 1) {
    if (tid < off) red[tid] += red[tid + off];
    __syncthreads();
  }
  if (tid == 0) out[b] = red[0] + ob[0];
}

extern "C" void kernel_launch(void* const* d_in, const int* in_sizes, int n_in,
                              void* d_out, int out_size, void* d_ws,
                              size_t ws_size, hipStream_t stream) {
  (void)in_sizes; (void)n_in; (void)out_size; (void)ws_size;
  const float* x    = (const float*)d_in[0];
  const float* ge   = (const float*)d_in[1];
  const float* cw   = (const float*)d_in[2];
  const float* cb   = (const float*)d_in[3];
  const float* wihf = (const float*)d_in[4];
  const float* bihf = (const float*)d_in[5];
  const float* bhhf = (const float*)d_in[6];
  const float* wihb = (const float*)d_in[7];
  const float* bihb = (const float*)d_in[8];
  const float* bhhb = (const float*)d_in[9];
  const float* gw   = (const float*)d_in[10];
  const float* gb   = (const float*)d_in[11];
  const float* gam  = (const float*)d_in[12];
  const float* bet  = (const float*)d_in[13];
  const float* lw   = (const float*)d_in[14];
  const float* lb   = (const float*)d_in[15];
  const float* ow   = (const float*)d_in[16];
  const float* ob   = (const float*)d_in[17];
  float* out = (float*)d_out;

  char* ws = (char*)d_ws;
  size_t off = 0;
  auto take = [&](size_t bytes) -> char* {
    char* p = ws + off;
    off = (off + bytes + 255) & ~(size_t)255;
    return p;
  };
  unsigned short* seq  = (unsigned short*)take(36000ull * 320 * 2);  // 23.0 MB
  float* h             = (float*)take(36000ull * 320 * 4);           // 46.1 MB
  float* gene_pre      = (float*)take(320ull * 800 * 4);
  float* gene_norm     = (float*)take(800ull * 320 * 4);
  unsigned short* flat = (unsigned short*)take(800ull * 14400 * 2);  // 23.0 MB
  float* hid           = (float*)take(800ull * 925 * 4);

  conv_pool_kernel<<<dim3(5, 3, 800), dim3(128), 0, stream>>>(x, cw, cb, seq);
  lstm_gates_kernel<<<dim3(10, 563, 2), dim3(128), 0, stream>>>(
      seq, wihf, bihf, bhhf, wihb, bihb, bhhb, h);
  gene_gemm_kernel<<<dim3(20, 13), dim3(128), 0, stream>>>(gw, ge, gb, gene_pre);
  bn_kernel<<<dim3(320), dim3(256), 0, stream>>>(gene_pre, gam, bet, gene_norm);
  attn_kernel<<<dim3(800, 45), dim3(128), 0, stream>>>(h, gene_norm, flat);
  lin_gemm_kernel<<<dim3(58, 13), dim3(128), 0, stream>>>(lw, flat, lb, hid);
  out_kernel<<<dim3(800), dim3(128), 0, stream>>>(hid, ow, ob, out);
}
